// CausalSelfAttention_8993661518424
// MI455X (gfx1250) — matrix operand order repaired and checked by reading, not compiled, not run on hardware
//
#include <hip/hip_runtime.h>

// Problem constants (B,T,C,H fixed by the reference).
namespace {
constexpr int B_  = 2;
constexpr int T_  = 2048;
constexpr int C_  = 1024;
constexpr int H_  = 16;
constexpr int HD_ = 64;           // C/H
constexpr int BT_ = B_ * T_;      // 4096
constexpr int C3_ = 3 * C_;       // 3072
}

typedef __attribute__((ext_vector_type(16))) _Float16     v16h;
typedef __attribute__((ext_vector_type(8)))  float        v8f;
typedef __attribute__((ext_vector_type(4)))  _Float16     v4h;
typedef __attribute__((ext_vector_type(4)))  unsigned int v4u;
typedef __attribute__((ext_vector_type(4)))  float        v4f;

// 16x16x32 f16 WMMA fragment: 16 halves / lane = 8 VGPRs.
union Frag {
  v16h v;
  v4u  q[2];
};

static __device__ __forceinline__ v8f wmma16(const Frag& a, const Frag& b, v8f c) {
  // 8 args: (neg_a, A, neg_b, B, c_mod, C, reuse_a, reuse_b)
  return __builtin_amdgcn_wmma_f32_16x16x32_f16(false, a.v, false, b.v, (short)0, c,
                                                false, false);
}

// ---------------------------------------------------------------------------
// 1) f32 -> f16 elementwise convert (vectorized x4)
// ---------------------------------------------------------------------------
__global__ __launch_bounds__(256) void k_cvt(const float* __restrict__ in,
                                             _Float16* __restrict__ out, int n4) {
  int i = blockIdx.x * 256 + threadIdx.x;
  if (i >= n4) return;
  v4f f = reinterpret_cast<const v4f*>(in)[i];
  v4h h;
  h.x = (_Float16)f.x; h.y = (_Float16)f.y;
  h.z = (_Float16)f.z; h.w = (_Float16)f.w;
  reinterpret_cast<v4h*>(out)[i] = h;
}

// ---------------------------------------------------------------------------
// 2) transpose f32 (R x Cc) -> f16 (Cc x R), tiled through LDS
// ---------------------------------------------------------------------------
__global__ __launch_bounds__(256) void k_transpose(const float* __restrict__ in,
                                                   _Float16* __restrict__ out,
                                                   int R, int Cc) {
  __shared__ float tile[32][33];
  const int c0 = blockIdx.x * 32, r0 = blockIdx.y * 32;
  const int tx = threadIdx.x, ty = threadIdx.y;
#pragma unroll
  for (int i = 0; i < 32; i += 8)
    tile[ty + i][tx] = in[(size_t)(r0 + ty + i) * Cc + (c0 + tx)];
  __syncthreads();
#pragma unroll
  for (int i = 0; i < 32; i += 8)
    out[(size_t)(c0 + ty + i) * R + (r0 + tx)] = (_Float16)tile[tx][ty + i];
}

// ---------------------------------------------------------------------------
// 3) QKV GEMM: (BT x C) @ (C x 3C) + bias, wave tile = 32(M) x 64(N)
//    (8 WMMAs per 12 x b128 loads per k-step -> higher matrix-op density).
//    Scatters outputs to head-major q (T x hd), k (T x hd), vT (hd x T).
// ---------------------------------------------------------------------------
__global__ __launch_bounds__(256) void k_qkv(const _Float16* __restrict__ xh,
                                             const _Float16* __restrict__ wt,   // (3C x C) K-major
                                             const float* __restrict__ bias,
                                             _Float16* __restrict__ qh,
                                             _Float16* __restrict__ kh,
                                             _Float16* __restrict__ vt) {
  const int lane = threadIdx.x & 31;
  const int wave = threadIdx.x >> 5;
  const int tile = blockIdx.x * 8 + wave;
  const int NT   = C3_ / 64;                 // 48 N-groups
  const int mt = tile / NT, nt = tile % NT;
  const int m0 = mt * 32, n0 = nt * 64;
  const int n  = lane & 15;
  const int hi = lane >> 4;

  v8f acc[2][4] = {};
  const _Float16* arow0 = xh + (size_t)(m0 + n) * C_ + hi * 8;
  const _Float16* arow1 = arow0 + (size_t)16 * C_;
  const _Float16* brow  = wt + (size_t)(n0 + n) * C_ + hi * 8;
  for (int k0 = 0; k0 < C_; k0 += 32) {
    __builtin_prefetch(arow0 + k0 + 64, 0, 0);
    __builtin_prefetch(arow1 + k0 + 64, 0, 0);
    Frag a0, a1;
    a0.q[0] = *reinterpret_cast<const v4u*>(arow0 + k0);
    a0.q[1] = *reinterpret_cast<const v4u*>(arow0 + k0 + 16);
    a1.q[0] = *reinterpret_cast<const v4u*>(arow1 + k0);
    a1.q[1] = *reinterpret_cast<const v4u*>(arow1 + k0 + 16);
#pragma unroll
    for (int j = 0; j < 4; ++j) {
      Frag b;
      const _Float16* p = brow + (size_t)j * 16 * C_ + k0;
      b.q[0] = *reinterpret_cast<const v4u*>(p);
      b.q[1] = *reinterpret_cast<const v4u*>(p + 16);
      acc[0][j] = wmma16(a0, b, acc[0][j]);
      acc[1][j] = wmma16(a1, b, acc[1][j]);
    }
  }

#pragma unroll
  for (int mh = 0; mh < 2; ++mh) {
    const int mrow = m0 + mh * 16;
#pragma unroll
    for (int j = 0; j < 4; ++j) {
      const int ng   = n0 + j * 16 + n;        // global column in [0,3C)
      const float bv = bias[ng];
      const int seg  = ng / C_;                // 0=q 1=k 2=v (wave-uniform)
      const int cc   = ng % C_;
      const int head = cc / HD_;
      const int d    = cc % HD_;
      if (seg == 2) {
        // vT[(b,h), d, t]: rows e are consecutive t -> one 16B store per lane
        alignas(16) _Float16 tmp[8];
#pragma unroll
        for (int e = 0; e < 8; ++e) tmp[e] = (_Float16)(acc[mh][j][e] + bv);
        const int bt0 = mrow + hi * 8;
        const int b = bt0 / T_, t0 = bt0 % T_;
        _Float16* dst = vt + ((size_t)(b * H_ + head) * HD_ + d) * T_ + t0;
        *reinterpret_cast<v4u*>(dst) = *reinterpret_cast<const v4u*>(tmp);
      } else {
        _Float16* base = (seg == 0) ? qh : kh;
#pragma unroll
        for (int e = 0; e < 8; ++e) {
          const int bt = mrow + hi * 8 + e;
          const int b = bt / T_, t = bt % T_;
          base[((size_t)(b * H_ + head) * T_ + t) * HD_ + d] =
              (_Float16)(acc[mh][j][e] + bv);
        }
      }
    }
  }
}

// ---------------------------------------------------------------------------
// 4) Flash attention: one wave per (b,h, 16-query tile); 32 keys / step.
// ---------------------------------------------------------------------------
__global__ __launch_bounds__(256) void k_attn(const _Float16* __restrict__ qh,
                                              const _Float16* __restrict__ kh,
                                              const _Float16* __restrict__ vt,
                                              _Float16* __restrict__ yh) {
  __shared__ __align__(16) _Float16 pbuf[8][16 * 32];   // per-wave P staging (1KB)
  const int lane = threadIdx.x & 31;
  const int wave = threadIdx.x >> 5;
  const int wid  = blockIdx.x * 8 + wave;   // 0 .. B*H*(T/16)-1
  const int QT   = T_ / 16;                 // 128
  const int bh   = wid / QT;
  const int m0   = (wid % QT) * 16;
  const int n    = lane & 15;
  const int hi   = lane >> 4;

  const _Float16* qb = qh + (size_t)bh * T_ * HD_;
  const _Float16* kb = kh + (size_t)bh * T_ * HD_;
  const _Float16* vb = vt + (size_t)bh * HD_ * T_;

  // Q fragments for hd=64 (two K=32 chunks), reused across all key steps.
  Frag aQ[2];
#pragma unroll
  for (int i = 0; i < 2; ++i) {
    const _Float16* p = qb + (size_t)(m0 + n) * HD_ + i * 32 + hi * 8;
    aQ[i].q[0] = *reinterpret_cast<const v4u*>(p);
    aQ[i].q[1] = *reinterpret_cast<const v4u*>(p + 16);
  }

  v8f accO[4] = {};
  float rowm[8], rowl[8];
#pragma unroll
  for (int e = 0; e < 8; ++e) { rowm[e] = -1e30f; rowl[e] = 0.f; }

  const float sc = 0.125f * 1.4426950408889634f;   // 1/sqrt(hd) * log2(e)
  _Float16* myp = pbuf[wave];

  for (int j0 = 0; j0 <= m0; j0 += 32) {
    // ---- S = Q K^T (two 16-key tiles) ----
    v8f s0 = {}, s1 = {};
#pragma unroll
    for (int t = 0; t < 2; ++t) {
      const int krl = j0 + 16 * t + n;
      const int kr  = (krl < T_ - 1) ? krl : (T_ - 1);   // clamp ragged tail
#pragma unroll
      for (int i = 0; i < 2; ++i) {
        Frag bk;
        const _Float16* p = kb + (size_t)kr * HD_ + i * 32 + hi * 8;
        bk.q[0] = *reinterpret_cast<const v4u*>(p);
        bk.q[1] = *reinterpret_cast<const v4u*>(p + 16);
        if (t == 0) s0 = wmma16(aQ[i], bk, s0);
        else        s1 = wmma16(aQ[i], bk, s1);
      }
    }

    // ---- causal mask (select kills NaN/garbage) + scale into exp2 domain ----
    float p0[8], p1[8], mx[8];
#pragma unroll
    for (int e = 0; e < 8; ++e) {
      const int q = m0 + e + hi * 8;
      p0[e] = (j0 + n      > q) ? -1e30f : s0[e] * sc;
      p1[e] = (j0 + 16 + n > q) ? -1e30f : s1[e] * sc;
      mx[e] = fmaxf(p0[e], p1[e]);
    }
    // row max across the 16-lane column group (xor<16 stays within halves)
#pragma unroll
    for (int e = 0; e < 8; ++e) {
      float m = mx[e];
      m = fmaxf(m, __shfl_xor(m, 1));
      m = fmaxf(m, __shfl_xor(m, 2));
      m = fmaxf(m, __shfl_xor(m, 4));
      m = fmaxf(m, __shfl_xor(m, 8));
      mx[e] = m;
    }
    float cf[8];
#pragma unroll
    for (int e = 0; e < 8; ++e) {
      const float nm = fmaxf(rowm[e], mx[e]);
      cf[e]   = exp2f(rowm[e] - nm);
      rowm[e] = nm;
      p0[e]   = exp2f(p0[e] - nm);
      p1[e]   = exp2f(p1[e] - nm);
      float ps = p0[e] + p1[e];
      ps += __shfl_xor(ps, 1);
      ps += __shfl_xor(ps, 2);
      ps += __shfl_xor(ps, 4);
      ps += __shfl_xor(ps, 8);
      rowl[e] = rowl[e] * cf[e] + ps;
    }
#pragma unroll
    for (int j = 0; j < 4; ++j)
#pragma unroll
      for (int e = 0; e < 8; ++e) accO[j][e] *= cf[e];

    // ---- stage P (C layout -> row-major 16x32 in LDS -> A layout) ----
#pragma unroll
    for (int e = 0; e < 8; ++e) {
      myp[(e + hi * 8) * 32 + n]      = (_Float16)p0[e];
      myp[(e + hi * 8) * 32 + 16 + n] = (_Float16)p1[e];
    }
    Frag aP;
    {
      const _Float16* p = myp + n * 32 + hi * 16;
      aP.q[0] = *reinterpret_cast<const v4u*>(p);
      aP.q[1] = *reinterpret_cast<const v4u*>(p + 8);
    }

    // ---- O += P V (vT rows are contiguous in t) ----
    const int c0l = j0 + hi * 16;
    const int c0  = (c0l < T_ - 16) ? c0l : (T_ - 16);   // clamp ragged tail
#pragma unroll
    for (int j = 0; j < 4; ++j) {
      Frag bv;
      const _Float16* p = vb + (size_t)(j * 16 + n) * T_ + c0;
      bv.q[0] = *reinterpret_cast<const v4u*>(p);
      bv.q[1] = *reinterpret_cast<const v4u*>(p + 8);
      accO[j] = wmma16(aP, bv, accO[j]);
    }
  }

  // ---- normalize and write y in (B*T, C) row-major f16 ----
  const int b = bh / H_, head = bh % H_;
  float inv[8];
#pragma unroll
  for (int e = 0; e < 8; ++e) inv[e] = 1.0f / rowl[e];
#pragma unroll
  for (int j = 0; j < 4; ++j)
#pragma unroll
    for (int e = 0; e < 8; ++e) {
      const int t = m0 + e + hi * 8;
      yh[((size_t)b * T_ + t) * C_ + head * HD_ + j * 16 + n] =
          (_Float16)(accO[j][e] * inv[e]);
    }
}

// ---------------------------------------------------------------------------
// 5) Output projection: (BT x C) @ (C x C) + bias -> f32 out,
//    wave tile = 32(M) x 64(N).
// ---------------------------------------------------------------------------
__global__ __launch_bounds__(256) void k_proj(const _Float16* __restrict__ yh,
                                              const _Float16* __restrict__ wt,  // (C x C) K-major
                                              const float* __restrict__ bias,
                                              float* __restrict__ out) {
  const int lane = threadIdx.x & 31;
  const int wave = threadIdx.x >> 5;
  const int tile = blockIdx.x * 8 + wave;
  const int NT   = C_ / 64;                 // 16
  const int mt = tile / NT, nt = tile % NT;
  const int m0 = mt * 32, n0 = nt * 64;
  const int n  = lane & 15;
  const int hi = lane >> 4;

  v8f acc[2][4] = {};
  const _Float16* arow0 = yh + (size_t)(m0 + n) * C_ + hi * 8;
  const _Float16* arow1 = arow0 + (size_t)16 * C_;
  const _Float16* brow  = wt + (size_t)(n0 + n) * C_ + hi * 8;
  for (int k0 = 0; k0 < C_; k0 += 32) {
    __builtin_prefetch(arow0 + k0 + 64, 0, 0);
    __builtin_prefetch(arow1 + k0 + 64, 0, 0);
    Frag a0, a1;
    a0.q[0] = *reinterpret_cast<const v4u*>(arow0 + k0);
    a0.q[1] = *reinterpret_cast<const v4u*>(arow0 + k0 + 16);
    a1.q[0] = *reinterpret_cast<const v4u*>(arow1 + k0);
    a1.q[1] = *reinterpret_cast<const v4u*>(arow1 + k0 + 16);
#pragma unroll
    for (int j = 0; j < 4; ++j) {
      Frag b;
      const _Float16* p = brow + (size_t)j * 16 * C_ + k0;
      b.q[0] = *reinterpret_cast<const v4u*>(p);
      b.q[1] = *reinterpret_cast<const v4u*>(p + 16);
      acc[0][j] = wmma16(a0, b, acc[0][j]);
      acc[1][j] = wmma16(a1, b, acc[1][j]);
    }
  }
#pragma unroll
  for (int mh = 0; mh < 2; ++mh) {
#pragma unroll
    for (int j = 0; j < 4; ++j) {
      const float bv = bias[n0 + j * 16 + n];
#pragma unroll
      for (int e = 0; e < 8; ++e)
        out[(size_t)(m0 + mh * 16 + e + hi * 8) * C_ + n0 + j * 16 + n] =
            acc[mh][j][e] + bv;
    }
  }
}

// ---------------------------------------------------------------------------
// Host launcher
// ---------------------------------------------------------------------------
extern "C" void kernel_launch(void* const* d_in, const int* in_sizes, int n_in,
                              void* d_out, int out_size, void* d_ws, size_t ws_size,
                              hipStream_t stream) {
  const float* x      = (const float*)d_in[0];
  const float* W_attn = (const float*)d_in[1];
  const float* b_attn = (const float*)d_in[2];
  const float* W_proj = (const float*)d_in[3];
  const float* b_proj = (const float*)d_in[4];
  float* out = (float*)d_out;

  // Workspace layout (f16): x | W_attn^T | W_proj^T | q | k | vT | y  = 48 MB
  char* ws = (char*)d_ws;
  size_t off = 0;
  _Float16* xh  = (_Float16*)(ws + off); off += (size_t)BT_ * C_ * 2;
  _Float16* wta = (_Float16*)(ws + off); off += (size_t)C3_ * C_ * 2;
  _Float16* wtp = (_Float16*)(ws + off); off += (size_t)C_  * C_ * 2;
  _Float16* qh  = (_Float16*)(ws + off); off += (size_t)BT_ * C_ * 2;
  _Float16* kh  = (_Float16*)(ws + off); off += (size_t)BT_ * C_ * 2;
  _Float16* vt  = (_Float16*)(ws + off); off += (size_t)BT_ * C_ * 2;
  _Float16* yh  = (_Float16*)(ws + off); off += (size_t)BT_ * C_ * 2;
  if (ws_size < off) return;   // insufficient scratch

  const int n4 = BT_ * C_ / 4;
  k_cvt<<<(n4 + 255) / 256, 256, 0, stream>>>(x, xh, n4);

  dim3 tb(32, 8);
  k_transpose<<<dim3(C3_ / 32, C_ / 32), tb, 0, stream>>>(W_attn, wta, C_, C3_);
  k_transpose<<<dim3(C_ / 32, C_ / 32), tb, 0, stream>>>(W_proj, wtp, C_, C_);

  // 128 M-tiles * 48 N-groups = 6144 waves, 8 waves/block
  k_qkv<<<(BT_ / 32) * (C3_ / 64) / 8, 256, 0, stream>>>(xh, wta, b_attn, qh, kh, vt);

  // B*H*(T/16) = 4096 waves
  k_attn<<<B_ * H_ * (T_ / 16) / 8, 256, 0, stream>>>(qh, kh, vt, yh);

  // 128 M-tiles * 16 N-groups = 2048 waves
  k_proj<<<(BT_ / 32) * (C_ / 64) / 8, 256, 0, stream>>>(yh, wtp, b_proj, out);
}
